// FusedTriangleMultiplicationOpti_77575699300523
// MI455X (gfx1250) — matrix operand order repaired and checked by reading, not compiled, not run on hardware
//
#include <hip/hip_runtime.h>
#include <hip/hip_bf16.h>

#define NN 512
#define CC 128
#define N2 (NN * NN)

typedef __attribute__((ext_vector_type(16))) __bf16 v16bf;
typedef __attribute__((ext_vector_type(8)))  __bf16 v8bf;
typedef __attribute__((ext_vector_type(8)))  float  v8f;
typedef __attribute__((ext_vector_type(4)))  float  v4f;

static __device__ __forceinline__ v16bf mk16(v8bf lo, v8bf hi) {
    return __builtin_shufflevector(lo, hi, 0,1,2,3,4,5,6,7,8,9,10,11,12,13,14,15);
}

static __device__ __forceinline__ float sigmoidf_(float x) {
    return 1.0f / (1.0f + __expf(-x));
}

// ---------------------------------------------------------------------------
// Kernel 0: f32 -> bf16 weight conversion
// ---------------------------------------------------------------------------
__global__ void k0_convert_bf16(const float* __restrict__ src,
                                __bf16* __restrict__ dst, int n) {
    int i = blockIdx.x * blockDim.x + threadIdx.x;
    if (i < n) dst[i] = (__bf16)src[i];
}

// ---------------------------------------------------------------------------
// Kernel 1: LN(act) -> [proj|gate|gl] GEMMs (WMMA bf16) -> mask*proj*sigmoid(gate)
//   writes leftT/rightT in (C, N, N) bf16, and sigmoid(gl) in (N*N, C) bf16
//   block: 256 threads (8 waves), handles 16 consecutive flat positions
// ---------------------------------------------------------------------------
__global__ __launch_bounds__(256) void k1_ln_proj(
    const float* __restrict__ act, const float* __restrict__ mask,
    const float* __restrict__ ln_w, const float* __restrict__ ln_b,
    const __bf16* __restrict__ proj_wb, const float* __restrict__ proj_b,
    const __bf16* __restrict__ gate_wb, const float* __restrict__ gate_b,
    const __bf16* __restrict__ gl_wb,   const float* __restrict__ gl_b,
    __bf16* __restrict__ leftT, __bf16* __restrict__ rightT,
    __bf16* __restrict__ gateout)
{
    __shared__ __bf16 xs[16][128];      // LN'd activations, bf16
    __shared__ __bf16 glbuf[16][128];   // sigmoid(gl) staging
    __shared__ float  ms[16];           // mask values
    __shared__ float  red[2][256];
    __shared__ float  stats[2][16];

    const int t    = threadIdx.x;
    const int pos0 = blockIdx.x * 16;

    // ---- Phase A: layernorm of 16 rows ----
    const int r   = t >> 4;
    const int seg = t & 15;
    const float* arow = act + (size_t)(pos0 + r) * CC + seg * 8;
    float v[8];
    float s = 0.f, sq = 0.f;
#pragma unroll
    for (int e = 0; e < 8; ++e) { v[e] = arow[e]; s += v[e]; sq += v[e] * v[e]; }
    red[0][t] = s; red[1][t] = sq;
    if (seg == 0) ms[r] = mask[pos0 + r];
    __syncthreads();
    if (seg == 0) {
        float S = 0.f, Q = 0.f;
#pragma unroll
        for (int k = 0; k < 16; ++k) { S += red[0][r * 16 + k]; Q += red[1][r * 16 + k]; }
        float mu  = S * (1.0f / CC);
        float var = Q * (1.0f / CC) - mu * mu;
        stats[0][r] = mu;
        stats[1][r] = __frsqrt_rn(var + 1e-5f);
    }
    __syncthreads();
    {
        float mu = stats[0][r], rs = stats[1][r];
#pragma unroll
        for (int e = 0; e < 8; ++e) {
            int c = seg * 8 + e;
            float x = (v[e] - mu) * rs * ln_w[c] + ln_b[c];
            xs[r][c] = (__bf16)x;
        }
    }
    __syncthreads();

    // ---- Phase B: WMMA GEMMs ----
    const int w      = t >> 5;          // wave id 0..7
    const int L      = t & 31;          // lane
    const int col    = L & 15;          // tile column (or A row)
    const int lhalf  = L >> 4;
    const int off1   = lhalf ? 16 : 0;  // A fragment K chunk offset
    const int off2   = lhalf ? 16 : 0;  // B fragment K offset
    const int rowoff = lhalf ? 8 : 0;   // C/D row offset

    // A fragments for all 4 K-steps (16x32 bf16 each), contiguous k = 16*lhalf + s
    v16bf afr[4];
#pragma unroll
    for (int ks = 0; ks < 4; ++ks) {
        int c0 = 32 * ks;
        v8bf lo = *(const v8bf*)&xs[col][c0 + off1];
        v8bf hi = *(const v8bf*)&xs[col][c0 + 8 + off1];
        afr[ks] = mk16(lo, hi);
    }

    const int d0a = 32 * w;            // proj/gate tile 0
    const int d0b = 32 * w + 16;       // proj/gate tile 1
    const int d0g = 16 * w;            // gl tile

    v8f z = {0.f,0.f,0.f,0.f,0.f,0.f,0.f,0.f};
    v8f p0 = z, p1 = z, g0 = z, g1 = z, gl = z;

#pragma unroll
    for (int ks = 0; ks < 4; ++ks) {
        int c0 = 32 * ks;
        v16bf b0 = *(const v16bf*)&proj_wb[(size_t)(d0a + col) * CC + c0 + off2];
        p0 = __builtin_amdgcn_wmma_f32_16x16x32_bf16(false, afr[ks], false, b0, (short)0, p0, false, false);
        v16bf b1 = *(const v16bf*)&proj_wb[(size_t)(d0b + col) * CC + c0 + off2];
        p1 = __builtin_amdgcn_wmma_f32_16x16x32_bf16(false, afr[ks], false, b1, (short)0, p1, false, false);
        v16bf b2 = *(const v16bf*)&gate_wb[(size_t)(d0a + col) * CC + c0 + off2];
        g0 = __builtin_amdgcn_wmma_f32_16x16x32_bf16(false, afr[ks], false, b2, (short)0, g0, false, false);
        v16bf b3 = *(const v16bf*)&gate_wb[(size_t)(d0b + col) * CC + c0 + off2];
        g1 = __builtin_amdgcn_wmma_f32_16x16x32_bf16(false, afr[ks], false, b3, (short)0, g1, false, false);
        v16bf b4 = *(const v16bf*)&gl_wb[(size_t)(d0g + col) * CC + c0 + off2];
        gl = __builtin_amdgcn_wmma_f32_16x16x32_bf16(false, afr[ks], false, b4, (short)0, gl, false, false);
    }

    // combine: mask * (proj + b) * sigmoid(gate + b); write transposed (C,N,N)
    {
        const float pb0 = proj_b[d0a + col], gb0 = gate_b[d0a + col];
        const float pb1 = proj_b[d0b + col], gb1 = gate_b[d0b + col];
        v8bf ov0, ov1;
#pragma unroll
        for (int vv = 0; vv < 8; ++vv) {
            int row = vv + rowoff;
            float m  = ms[row];
            float pa = m * (p0[vv] + pb0) * sigmoidf_(g0[vv] + gb0);
            float pc = m * (p1[vv] + pb1) * sigmoidf_(g1[vv] + gb1);
            ov0[vv] = (__bf16)pa;
            ov1[vv] = (__bf16)pc;
        }
        if (w < 4) {  // d0a,d0b in [0,128): left half
            *(v8bf*)&leftT[(size_t)(d0a + col) * N2 + pos0 + rowoff] = ov0;
            *(v8bf*)&leftT[(size_t)(d0b + col) * N2 + pos0 + rowoff] = ov1;
        } else {      // right half
            *(v8bf*)&rightT[(size_t)(d0a + col - CC) * N2 + pos0 + rowoff] = ov0;
            *(v8bf*)&rightT[(size_t)(d0b + col - CC) * N2 + pos0 + rowoff] = ov1;
        }
        // gl sigmoid -> staging LDS (row-major), then coalesced store
        const float glb = gl_b[d0g + col];
#pragma unroll
        for (int vv = 0; vv < 8; ++vv) {
            int row = vv + rowoff;
            glbuf[row][d0g + col] = (__bf16)sigmoidf_(gl[vv] + glb);
        }
    }
    __syncthreads();
    {
        int row = t >> 4, ch = t & 15;
        *(v8bf*)&gateout[(size_t)(pos0 + row) * CC + ch * 8] = *(v8bf*)&glbuf[row][ch * 8];
    }
}

// ---------------------------------------------------------------------------
// Kernel 2: out[i,j,c] = sum_k left[i,k,c]*right[j,k,c]  (128 batched matmuls)
//   + fused channel-LayerNorm, writes LN'd result as bf16 (N*N, C)
//   block: 512 threads (16 waves); workgroup tile = 16(i) x 32(j) x all 128 ch
//   wave = 8 channels x 2 j-tiles  (A fragment reused across both j-tiles)
//   dynamic LDS: 512 positions x 132 floats (padded) = 270 KB  (CDNA5 big LDS)
// ---------------------------------------------------------------------------
__global__ __launch_bounds__(512) void k2_pair(
    const __bf16* __restrict__ leftT, const __bf16* __restrict__ rightT,
    const float* __restrict__ cn_w, const float* __restrict__ cn_b,
    __bf16* __restrict__ lnout)
{
    extern __shared__ float smem[];     // [512][132]

    const int t     = threadIdx.x;
    const int w     = t >> 5;           // wave 0..15: channels [8w, 8w+8)
    const int L     = t & 31;
    const int col   = L & 15;
    const int lhalf = L >> 4;
    const int off1  = lhalf ? 16 : 0;
    const int off2  = lhalf ? 16 : 0;
    const int rowoff= lhalf ? 8 : 0;

    const int i0 = blockIdx.y * 16;
    const int j0 = blockIdx.x * 32;

    v8f z = {0.f,0.f,0.f,0.f,0.f,0.f,0.f,0.f};
    v8f acc[8][2];
#pragma unroll
    for (int cc = 0; cc < 8; ++cc) { acc[cc][0] = z; acc[cc][1] = z; }

    const __bf16* aBase  = leftT  + (size_t)(i0 + col) * NN;
    const __bf16* bBase0 = rightT + (size_t)(j0 + col) * NN;
    const __bf16* bBase1 = rightT + (size_t)(j0 + 16 + col) * NN;

    for (int ks = 0; ks < 16; ++ks) {
        const int k0 = ks * 32;
#pragma unroll
        for (int cc = 0; cc < 8; ++cc) {
            const size_t coff = (size_t)(w * 8 + cc) * N2;
            v8bf lo = *(const v8bf*)(aBase + coff + k0 + off1);
            v8bf hi = *(const v8bf*)(aBase + coff + k0 + 8 + off1);
            v16bf a  = mk16(lo, hi);
            v16bf b0 = *(const v16bf*)(bBase0 + coff + k0 + off2);
            v16bf b1 = *(const v16bf*)(bBase1 + coff + k0 + off2);
            acc[cc][0] = __builtin_amdgcn_wmma_f32_16x16x32_bf16(false, a, false, b0, (short)0, acc[cc][0], false, false);
            acc[cc][1] = __builtin_amdgcn_wmma_f32_16x16x32_bf16(false, a, false, b1, (short)0, acc[cc][1], false, false);
        }
    }

    // spill 16x32x128 f32 tile to LDS: smem[pos][c], pos = Mrow*32 + Ncol
#pragma unroll
    for (int cc = 0; cc < 8; ++cc) {
        int c = w * 8 + cc;
#pragma unroll
        for (int jt = 0; jt < 2; ++jt) {
#pragma unroll
            for (int vv = 0; vv < 8; ++vv) {
                int p = (vv + rowoff) * 32 + jt * 16 + col;
                smem[p * 132 + c] = acc[cc][jt][vv];
            }
        }
    }
    __syncthreads();

    // channel layernorm per position + bf16 output, coalesced rows
    {
        const int p = t;                // 512 threads <-> 512 positions
        const float* base = smem + p * 132;
        float S = 0.f, Q = 0.f;
        const v4f* b4 = (const v4f*)base;
#pragma unroll
        for (int q = 0; q < 32; ++q) {
            v4f x = b4[q];
#pragma unroll
            for (int e = 0; e < 4; ++e) { S += x[e]; Q += x[e] * x[e]; }
        }
        float mu  = S * (1.0f / CC);
        float var = Q * (1.0f / CC) - mu * mu;
        float rs  = __frsqrt_rn(var + 1e-5f);

        const int Mi = p >> 5, Nj = p & 31;
        __bf16* orow = lnout + ((size_t)(i0 + Mi) * NN + (j0 + Nj)) * CC;
#pragma unroll
        for (int c0 = 0; c0 < 128; c0 += 8) {
            v8bf ov;
#pragma unroll
            for (int e = 0; e < 8; ++e) {
                int c = c0 + e;
                ov[e] = (__bf16)((base[c] - mu) * rs * cn_w[c] + cn_b[c]);
            }
            *(v8bf*)&orow[c0] = ov;
        }
    }
}

// ---------------------------------------------------------------------------
// Kernel 3: final projection (WMMA) + bias + multiply by stored sigmoid gate
//   block: 256 threads (8 waves), 16 rows, one 16-col tile per wave
// ---------------------------------------------------------------------------
__global__ __launch_bounds__(256) void k3_out(
    const __bf16* __restrict__ lnout, const __bf16* __restrict__ gateout,
    const __bf16* __restrict__ outp_wb, const float* __restrict__ outp_b,
    float* __restrict__ out)
{
    __shared__ __bf16 xs[16][128];
    __shared__ __bf16 gs[16][128];
    __shared__ float  ob[16][132];

    const int t    = threadIdx.x;
    const int pos0 = blockIdx.x * 16;

    {   // cooperative loads (coalesced 16B chunks)
        int row = t >> 4, ch = t & 15;
        *(v8bf*)&xs[row][ch * 8] = *(const v8bf*)&lnout[(size_t)(pos0 + row) * CC + ch * 8];
        *(v8bf*)&gs[row][ch * 8] = *(const v8bf*)&gateout[(size_t)(pos0 + row) * CC + ch * 8];
    }
    __syncthreads();

    const int w      = t >> 5;
    const int L      = t & 31;
    const int col    = L & 15;
    const int lhalf  = L >> 4;
    const int off1   = lhalf ? 16 : 0;
    const int off2   = lhalf ? 16 : 0;
    const int rowoff = lhalf ? 8 : 0;

    v8f acc = {0.f,0.f,0.f,0.f,0.f,0.f,0.f,0.f};
    const int d = 16 * w + col;
#pragma unroll
    for (int ks = 0; ks < 4; ++ks) {
        int c0 = 32 * ks;
        v8bf lo = *(const v8bf*)&xs[col][c0 + off1];
        v8bf hi = *(const v8bf*)&xs[col][c0 + 8 + off1];
        v16bf a = mk16(lo, hi);
        v16bf b = *(const v16bf*)&outp_wb[(size_t)d * CC + c0 + off2];
        acc = __builtin_amdgcn_wmma_f32_16x16x32_bf16(false, a, false, b, (short)0, acc, false, false);
    }

    const float bias = outp_b[d];
#pragma unroll
    for (int vv = 0; vv < 8; ++vv) {
        int row = vv + rowoff;
        float g = (float)gs[row][d];
        ob[row][d] = (acc[vv] + bias) * g;
    }
    __syncthreads();

    {   // coalesced 16B f32 stores
        int row = t >> 4, ch = t & 15;
        *(v4f*)&out[(size_t)(pos0 + row) * CC + ch * 8]     = *(const v4f*)&ob[row][ch * 8];
        *(v4f*)&out[(size_t)(pos0 + row) * CC + ch * 8 + 4] = *(const v4f*)&ob[row][ch * 8 + 4];
    }
}

// ---------------------------------------------------------------------------
extern "C" void kernel_launch(void* const* d_in, const int* in_sizes, int n_in,
                              void* d_out, int out_size, void* d_ws, size_t ws_size,
                              hipStream_t stream) {
    const float* act    = (const float*)d_in[0];
    const float* mask   = (const float*)d_in[1];
    const float* ln_w   = (const float*)d_in[2];
    const float* ln_b   = (const float*)d_in[3];
    const float* proj_w = (const float*)d_in[4];
    const float* proj_b = (const float*)d_in[5];
    const float* gate_w = (const float*)d_in[6];
    const float* gate_b = (const float*)d_in[7];
    const float* cn_w   = (const float*)d_in[8];
    const float* cn_b   = (const float*)d_in[9];
    const float* outp_w = (const float*)d_in[10];
    const float* outp_b = (const float*)d_in[11];
    const float* gl_w   = (const float*)d_in[12];
    const float* gl_b   = (const float*)d_in[13];
    float* out = (float*)d_out;

    // workspace layout (bf16 elements)
    __bf16* leftT   = (__bf16*)d_ws;                    // 128*N2
    __bf16* rightT  = leftT   + (size_t)CC * N2;        // 128*N2
    __bf16* gateout = rightT  + (size_t)CC * N2;        // N2*128
    __bf16* lnout   = gateout + (size_t)N2 * CC;        // N2*128
    __bf16* proj_wb = lnout   + (size_t)N2 * CC;        // 256*128
    __bf16* gate_wb = proj_wb + 2 * CC * CC;            // 256*128
    __bf16* outp_wb = gate_wb + 2 * CC * CC;            // 128*128
    __bf16* gl_wb   = outp_wb + CC * CC;                // 128*128

    // kernel 0: weight conversions
    k0_convert_bf16<<<(2 * CC * CC + 255) / 256, 256, 0, stream>>>(proj_w, proj_wb, 2 * CC * CC);
    k0_convert_bf16<<<(2 * CC * CC + 255) / 256, 256, 0, stream>>>(gate_w, gate_wb, 2 * CC * CC);
    k0_convert_bf16<<<(CC * CC + 255) / 256, 256, 0, stream>>>(outp_w, outp_wb, CC * CC);
    k0_convert_bf16<<<(CC * CC + 255) / 256, 256, 0, stream>>>(gl_w, gl_wb, CC * CC);

    // kernel 1: LN + proj/gate/gl GEMMs, fused mask+sigmoid, transposed store
    k1_ln_proj<<<N2 / 16, 256, 0, stream>>>(act, mask, ln_w, ln_b,
                                            proj_wb, proj_b, gate_wb, gate_b,
                                            gl_wb, gl_b, leftT, rightT, gateout);

    // kernel 2: 128 batched 512^3 matmuls + fused channel-LN (270 KB dyn LDS)
    k2_pair<<<dim3(NN / 32, NN / 16), 512, 512 * 132 * sizeof(float), stream>>>(
        leftT, rightT, cn_w, cn_b, lnout);

    // kernel 3: output projection + sigmoid gating
    k3_out<<<N2 / 16, 256, 0, stream>>>(lnout, gateout, outp_wb, outp_b, out);
}
